// ConvNd_12781822673276
// MI455X (gfx1250) — matrix operand order repaired and checked by reading, not compiled, not run on hardware
//
#include <hip/hip_runtime.h>

// ---------------------------------------------------------------------------
// 4D conv (k=3^4, pad=1, stride=1) as implicit GEMM on CDNA5 bf16 WMMA.
//   M = O (64), N = output positions, K = Ci*81 = 2592.
// Compute-bound (~220 GFLOP vs ~255MB traffic) -> v_wmma_f32_16x16x32_bf16.
// Weights staged to LDS per (k0,k1) slice, double-buffered, via
// global_load_async_to_lds_b128 (ASYNCcnt); B fragments preloaded per slice
// to avoid per-WMMA s_wait_loadcnt 0 serialization.
// d_ws layout: [0, 58,492,928)        padded channels-last bf16 x (2,26^4,32)
//              [58,492,928, +331,776) weights in WMMA A-fragment order (bf16)
// ---------------------------------------------------------------------------

typedef __attribute__((ext_vector_type(16))) __bf16 v16bf;
typedef __attribute__((ext_vector_type(8)))  float  v8f;
typedef __attribute__((ext_vector_type(4)))  int    v4i;

constexpr int  kD  = 24;       // spatial dim (all four)
constexpr int  kP  = 26;       // padded spatial dim
constexpr int  kCI = 32;       // input channels
constexpr int  kCO = 64;       // output channels
constexpr int  kTAPS = 81;     // 3^4 kernel taps
constexpr long long kSP   = (long long)kD * kD * kD * kD;      // 331776
constexpr long long kNPOS = 2LL * kP * kP * kP * kP;           // padded positions

constexpr int kSliceTaps  = 9;                      // (k2,k3) taps per (k0,k1) slice
constexpr int kSliceElems = kSliceTaps * 2048;      // 18432 bf16 = 36864 B
constexpr int kSliceVec16 = kSliceTaps * 4096 / 16; // 2304 x 16B chunks

#if __has_builtin(__builtin_amdgcn_global_load_async_to_lds_b128) && \
    __has_builtin(__builtin_amdgcn_s_wait_asynccnt)
#define USE_ASYNC_LDS 1
#else
#define USE_ASYNC_LDS 0
#endif

static __device__ __forceinline__ unsigned short f32_to_bf16(float f) {
  union { float f; unsigned int u; } v; v.f = f;
  unsigned int u = v.u;
  unsigned int r = u + 0x7FFFu + ((u >> 16) & 1u);   // round-to-nearest-even
  return (unsigned short)(r >> 16);
}

// ---- pre-pass 1: x (N,Ci,24^4) f32 -> xT (N,26,26,26,26,Ci) bf16, zero halo ----
__global__ void xpad_kernel(const float* __restrict__ x,
                            unsigned short* __restrict__ xT) {
  long long pos = (long long)blockIdx.x * blockDim.x + threadIdx.x;
  if (pos >= kNPOS) return;
  long long t = pos;
  int p3 = (int)(t % kP); t /= kP;
  int p2 = (int)(t % kP); t /= kP;
  int p1 = (int)(t % kP); t /= kP;
  int p0 = (int)(t % kP); int n = (int)(t / kP);
  unsigned short* dst = xT + pos * kCI;
  bool inb = (p0 >= 1 && p0 <= kD) && (p1 >= 1 && p1 <= kD) &&
             (p2 >= 1 && p2 <= kD) && (p3 >= 1 && p3 <= kD);
  if (!inb) {
#pragma unroll
    for (int c = 0; c < kCI; ++c) dst[c] = 0;
    return;
  }
  long long sbase = ((((long long)n * kCI) * kD + (p0 - 1)) * kD + (p1 - 1)) * kD * kD
                    + (long long)(p2 - 1) * kD + (p3 - 1);
#pragma unroll
  for (int c = 0; c < kCI; ++c)
    dst[c] = f32_to_bf16(x[sbase + (long long)c * kSP]);
}

// ---- pre-pass 2: w (3,64,32,3,3,3) f32 -> w2[tap][otile][lane][16] bf16 -------
// A-fragment order (contiguous k, same as B): lane m   (m<16): K = {0..15}
//                                        lane m+16      : K = {16..31}
__global__ void wprep_kernel(const float* __restrict__ w,
                             unsigned short* __restrict__ w2) {
  int idx = blockIdx.x * blockDim.x + threadIdx.x;   // 81*4*32*16 = 165888
  if (idx >= kTAPS * 4 * 32 * 16) return;
  int e    = idx & 15;
  int lane = (idx >> 4) & 31;
  int ot   = (idx >> 9) & 3;
  int tap  = idx >> 11;
  int m = lane & 15, half = lane >> 4;
  int o  = ot * 16 + m;
  int ci = e + half * 16;
  int k3 = tap % 3, k2 = (tap / 3) % 3, k1 = (tap / 9) % 3, k0 = tap / 27;
  long long widx = (((long long)k0 * kCO + o) * kCI + ci) * 27 + (k1 * 9 + k2 * 3 + k3);
  w2[idx] = f32_to_bf16(w[widx]);
}

// ---- cooperative weight-slice staging: global (contiguous) -> LDS buffer -----
static __device__ __forceinline__ void stage_slice(const unsigned short* __restrict__ w2,
                                                   int slice,
                                                   unsigned short* lds_buf,
                                                   int tid) {
  const unsigned short* src = w2 + (long long)slice * kSliceElems;
#if USE_ASYNC_LDS
  for (int i = tid; i < kSliceVec16; i += 128) {
    __builtin_amdgcn_global_load_async_to_lds_b128(
        (__attribute__((address_space(1))) v4i*)(src + i * 8),
        (__attribute__((address_space(3))) v4i*)(lds_buf + i * 8),
        0, 0);
  }
#else
  for (int i = tid; i < kSliceVec16; i += 128) {
    uint4 tmp = *(const uint4*)(src + i * 8);
    *(uint4*)(lds_buf + i * 8) = tmp;
  }
#endif
}

// ---- main: one wave per 16-position tile, all 64 output channels --------------
__global__ void __launch_bounds__(128)
conv4d_wmma_kernel(const unsigned short* __restrict__ xT,
                   const unsigned short* __restrict__ w2,
                   const float* __restrict__ bias,
                   float* __restrict__ out) {
  __shared__ __align__(16) unsigned short lds_w[2 * kSliceElems];  // 73728 B

  const int tid  = threadIdx.x;
  const int lane = tid & 31;
  const int wave = tid >> 5;
  int tileId = blockIdx.x * 4 + wave;            // 41472 tiles total
  int t   = tileId % 36; int rest = tileId / 36; // 36 tiles per (n,d0,d1)
  int d1  = rest % kD;  rest /= kD;
  int d0  = rest % kD;  int n = rest / kD;
  int half = lane >> 4;
  int col  = lane & 15;
  int p    = t * 16 + col;                       // flattened (d2,d3) position
  int d2   = p / kD, d3 = p % kD;

  // B fragment base (channels-last padded x); lane half selects ci 0-15 / 16-31
  const unsigned short* xbase =
      xT + (((((long long)n * kP + d0) * kP + d1) * kP + d2) * kP + d3) * kCI + half * 16;

  // prologue: stage slice 0
  stage_slice(w2, 0, lds_w, tid);
#if USE_ASYNC_LDS
  __builtin_amdgcn_s_wait_asynccnt(0);
#endif
  __syncthreads();

  v8f acc0 = {}, acc1 = {}, acc2 = {}, acc3 = {};

  for (int s = 0; s < 9; ++s) {                  // slice = (k0,k1)
    const int k0 = s / 3, k1 = s % 3;

    // kick off async copy of next slice into the other LDS buffer
    if (s + 1 < 9)
      stage_slice(w2, s + 1, lds_w + ((s + 1) & 1) * kSliceElems, tid);

    const unsigned short* wl = lds_w + (s & 1) * kSliceElems + lane * 16;
    const unsigned short* xs = xbase + (((long long)k0 * kP + k1) * kP) * kP * kCI;

    // preload all 9 B fragments of this slice (one clause of global_load_b128)
    v16bf Bf[kSliceTaps];
#pragma unroll
    for (int tl = 0; tl < kSliceTaps; ++tl) {
      const int k2 = tl / 3, k3 = tl % 3;
      Bf[tl] = *(const v16bf*)(xs + ((long long)k2 * kP + k3) * kCI);
    }

#pragma unroll
    for (int tl = 0; tl < kSliceTaps; ++tl) {
      v16bf A0 = *(const v16bf*)(wl + tl * 2048 + 0 * 512);
      v16bf A1 = *(const v16bf*)(wl + tl * 2048 + 1 * 512);
      v16bf A2 = *(const v16bf*)(wl + tl * 2048 + 2 * 512);
      v16bf A3 = *(const v16bf*)(wl + tl * 2048 + 3 * 512);
      acc0 = __builtin_amdgcn_wmma_f32_16x16x32_bf16(false, A0, false, Bf[tl], (short)0, acc0, false, false);
      acc1 = __builtin_amdgcn_wmma_f32_16x16x32_bf16(false, A1, false, Bf[tl], (short)0, acc1, false, false);
      acc2 = __builtin_amdgcn_wmma_f32_16x16x32_bf16(false, A2, false, Bf[tl], (short)0, acc2, false, false);
      acc3 = __builtin_amdgcn_wmma_f32_16x16x32_bf16(false, A3, false, Bf[tl], (short)0, acc3, false, false);
    }

    if (s + 1 < 9) {
#if USE_ASYNC_LDS
      __builtin_amdgcn_s_wait_asynccnt(0);
#endif
      __syncthreads();
    }
  }

  // C/D layout: VGPR r -> row M = r (lanes 0-15) or r+8 (lanes 16-31), col N = lane%16
  long long obase = ((long long)n * kCO) * kSP +
                    (((long long)d0 * kD + d1) * kD + d2) * kD + d3;
#pragma unroll
  for (int ot = 0; ot < 4; ++ot) {
    v8f a = (ot == 0) ? acc0 : (ot == 1) ? acc1 : (ot == 2) ? acc2 : acc3;
#pragma unroll
    for (int r = 0; r < 8; ++r) {
      int o = ot * 16 + half * 8 + r;
      out[obase + (long long)o * kSP] = a[r] + bias[o];
    }
  }
}

extern "C" void kernel_launch(void* const* d_in, const int* in_sizes, int n_in,
                              void* d_out, int out_size, void* d_ws, size_t ws_size,
                              hipStream_t stream) {
  const float* x = (const float*)d_in[0];   // (2,32,24,24,24,24)
  const float* w = (const float*)d_in[1];   // (3,64,32,3,3,3)
  const float* b = (const float*)d_in[2];   // (64,)
  float* out = (float*)d_out;               // (2,64,24,24,24,24)

  unsigned short* xT = (unsigned short*)d_ws;           // kNPOS*kCI bf16
  unsigned short* w2 = xT + kNPOS * kCI;                // 165888 bf16

  int xblocks = (int)((kNPOS + 255) / 256);
  xpad_kernel<<<xblocks, 256, 0, stream>>>(x, xT);
  wprep_kernel<<<(kTAPS * 4 * 32 * 16 + 255) / 256, 256, 0, stream>>>(w, w2);

  // 2*24*24*36 = 41472 tiles, 4 waves (tiles) per 128-thread block
  conv4d_wmma_kernel<<<41472 / 4, 128, 0, stream>>>(xT, w2, b, out);
}
